// BoundaryPredictor2_66975720014433
// MI455X (gfx1250) — matrix operand order repaired and checked by reading, not compiled, not run on hardware
//
#include <hip/hip_runtime.h>
#include <hip/hip_bf16.h>
#include <math.h>

#define NB 8
#define NL 2048
#define ND 512
#define APAD 520   // padded LDS row stride (halves) for sA (bank-conflict avoidance)
#define NPAD 520   // padded LDS row stride (halves) for sB

typedef __attribute__((ext_vector_type(16))) _Float16     v16h;
typedef __attribute__((ext_vector_type(8)))  float        v8f;
typedef __attribute__((ext_vector_type(4)))  float        v4f;
typedef __attribute__((ext_vector_type(4)))  unsigned int v4u;
typedef __attribute__((ext_vector_type(2)))  unsigned int v2u;

union Frag16 { v16h v; v4u q[2]; };
union H4     { _Float16 h[4]; v2u u2; };

__device__ __forceinline__ float wave_sum32(float v) {
#pragma unroll
    for (int m = 16; m >= 1; m >>= 1) v += __shfl_xor(v, m, 32);
    return v;
}

// -------------------------------------------------------------------------
// Kernel 1: fused normalize + Q/K GEMM (WMMA f16) + row-wise dot -> cos_sim
// Block = 256 threads (8 waves); one block per (b, 16-token tile).
// A-layout (ISA 16-bit A 16x32): lane -> M=lane&15, baseK=(lane<16)?0:8,
//   halves = two contiguous 8-half runs at K=baseK and K=baseK+16.
// B-layout: lane -> N=lane&15, halves -> K in the same order as A (strided NPAD reads from K-major sB).
// -------------------------------------------------------------------------
__global__ __launch_bounds__(256) void k_gemm_cos(
    const float* __restrict__ hidden,
    const float* __restrict__ Wq,
    const float* __restrict__ Wk,
    float* __restrict__ cos_out)
{
    __shared__ _Float16 sA[17 * APAD];   // 17 normalized rows (17680 B)
    __shared__ _Float16 sB[32 * NPAD];   // 32(k) x 512(n) slab, K-major (33280 B)
    __shared__ float    sWD[8][16];

    const int b    = blockIdx.y;
    const int l0   = blockIdx.x * 16;
    const int tid  = threadIdx.x;
    const int wave = tid >> 5;
    const int lane = tid & 31;

    // ---- normalize rows l0..l0+16 into sA (f16), float4 path ----
    for (int r = wave; r < 17; r += 8) {
        int row = l0 + r; if (row > NL - 1) row = NL - 1;
        const v4f* src4 = (const v4f*)(hidden + ((size_t)b * NL + row) * ND);
        v4f vv[4];
        float ss = 0.0f;
#pragma unroll
        for (int i = 0; i < 4; ++i) {
            vv[i] = src4[lane + 32 * i];
            ss += vv[i][0]*vv[i][0] + vv[i][1]*vv[i][1] + vv[i][2]*vv[i][2] + vv[i][3]*vv[i][3];
        }
        ss = wave_sum32(ss);
        float nrm = sqrtf(ss);
        if (nrm < 1e-12f) nrm = 1e-12f;
        float inv = 1.0f / nrm;
#pragma unroll
        for (int i = 0; i < 4; ++i) {
            H4 h4;
#pragma unroll
            for (int j = 0; j < 4; ++j) h4.h[j] = (_Float16)(vv[i][j] * inv);
            *(v2u*)(sA + r * APAD + (lane + 32 * i) * 4) = h4.u2;
        }
    }

    const int mrow  = lane & 15;
    const int baseK = (lane < 16) ? 0 : 8;

    v8f accQ[4], accK[4];
    const v8f vzero = {0.f,0.f,0.f,0.f,0.f,0.f,0.f,0.f};
#pragma unroll
    for (int j = 0; j < 4; ++j) { accQ[j] = vzero; accK[j] = vzero; }

    const int kk = lane;          // slab k-index this thread stages
    const int nq = wave;          // n-quad group base

    for (int ks = 0; ks < 16; ++ks) {
        const int k0 = ks * 32;

        // ---- stage Wq slab (K-major): sB[kk*NPAD + n] = Wq[n][k0+kk] ----
        __syncthreads();
#pragma unroll
        for (int i = 0; i < 16; ++i) {
            int nbase = (nq + 8 * i) * 4;
            H4 h4;
#pragma unroll
            for (int j = 0; j < 4; ++j)
                h4.h[j] = (_Float16)Wq[(size_t)(nbase + j) * ND + k0 + kk];
            *(v2u*)(sB + kk * NPAD + nbase) = h4.u2;
        }
        __syncthreads();

        Frag16 aq;
        aq.q[0] = *(const v4u*)(sA + mrow * APAD + k0 + baseK);
        aq.q[1] = *(const v4u*)(sA + mrow * APAD + k0 + baseK + 16);
#pragma unroll
        for (int j = 0; j < 4; ++j) {
            int n0 = (wave + 8 * j) * 16;
            Frag16 bf;
            for (int s = 0; s < 16; ++s)
                bf.v[s] = sB[(baseK + (s & 7) + ((s >> 3) << 4)) * NPAD + n0 + mrow];
            accQ[j] = __builtin_amdgcn_wmma_f32_16x16x32_f16(
                false, aq.v, false, bf.v, (short)0, accQ[j], false, false);
        }
        __syncthreads();

        // ---- stage Wk slab ----
#pragma unroll
        for (int i = 0; i < 16; ++i) {
            int nbase = (nq + 8 * i) * 4;
            H4 h4;
#pragma unroll
            for (int j = 0; j < 4; ++j)
                h4.h[j] = (_Float16)Wk[(size_t)(nbase + j) * ND + k0 + kk];
            *(v2u*)(sB + kk * NPAD + nbase) = h4.u2;
        }
        __syncthreads();

        Frag16 ak;                 // rows 1..16 of the tile
        ak.q[0] = *(const v4u*)(sA + (mrow + 1) * APAD + k0 + baseK);
        ak.q[1] = *(const v4u*)(sA + (mrow + 1) * APAD + k0 + baseK + 16);
#pragma unroll
        for (int j = 0; j < 4; ++j) {
            int n0 = (wave + 8 * j) * 16;
            Frag16 bf;
            for (int s = 0; s < 16; ++s)
                bf.v[s] = sB[(baseK + (s & 7) + ((s >> 3) << 4)) * NPAD + n0 + mrow];
            accK[j] = __builtin_amdgcn_wmma_f32_16x16x32_f16(
                false, ak.v, false, bf.v, (short)0, accK[j], false, false);
        }
    }

    // ---- dot: C layout lane -> N=lane&15, vgpr r -> M = r + 8*(lane>=16) ----
    float part[8];
#pragma unroll
    for (int r = 0; r < 8; ++r) {
        float s = 0.0f;
#pragma unroll
        for (int j = 0; j < 4; ++j) s += accQ[j][r] * accK[j][r];
        part[r] = s;
    }
#pragma unroll
    for (int m = 8; m >= 1; m >>= 1)
#pragma unroll
        for (int r = 0; r < 8; ++r) part[r] += __shfl_xor(part[r], m, 32);

    if (lane == 0) {
#pragma unroll
        for (int r = 0; r < 8; ++r) sWD[wave][r] = part[r];
    }
    if (lane == 16) {
#pragma unroll
        for (int r = 0; r < 8; ++r) sWD[wave][8 + r] = part[r];
    }
    __syncthreads();
    if (tid < 16) {
        int l = l0 + tid;
        if (l < NL - 1) {
            float s = 0.0f;
#pragma unroll
            for (int w = 0; w < 8; ++w) s += sWD[w][tid];
            cos_out[b * NL + l] = s;
        }
    }
}

// -------------------------------------------------------------------------
// Kernel 2: per-batch boundary logic, prefix scan, stats. 1 block per batch.
// -------------------------------------------------------------------------
#define TPT 8  // tokens per thread (256 * 8 = 2048)
__global__ __launch_bounds__(256) void k_boundaries(
    const float* __restrict__ cosv, const float* __restrict__ amask,
    const float* __restrict__ noise,
    float* __restrict__ out_sm, float* __restrict__ out_lp,
    float* __restrict__ out_conf, float* __restrict__ out_ent,
    int* __restrict__ bpos, int* __restrict__ counts_ws, int* __restrict__ len_ws)
{
    __shared__ float spart[8];
    __shared__ int   sscan[257];
    __shared__ int   s_len;

    const int b    = blockIdx.x;
    const int tid  = threadIdx.x;
    const int wave = tid >> 5;
    const int lane = tid & 31;
    const int l0   = tid * TPT;

    // ---- sequence length ----
    float msum = 0.0f;
    for (int i = 0; i < TPT; ++i) msum += amask[b * NL + l0 + i];
    msum = wave_sum32(msum);
    if (lane == 0) spart[wave] = msum;
    __syncthreads();
    if (tid == 0) {
        float s = 0;
#pragma unroll
        for (int w = 0; w < 8; ++w) s += spart[w];
        s_len = (int)(s + 0.5f);
    }
    __syncthreads();
    const int len = s_len;

    // ---- per-token boundary + stats ----
    const float EPSf = 1.1920929e-07f;
    float lp = 0.0f, conf = 0.0f, ent = 0.0f;
    int flags = 0, cnt = 0;
    for (int i = 0; i < TPT; ++i) {
        int l = l0 + i;
        float p;
        if (l < NL - 1) {
            float c = cosv[b * NL + l];
            p = (1.0f - c) * 0.5f;
            p = fminf(fmaxf(p, 0.0f), 1.0f);
        } else p = 0.0f;
        float m = (l < len) ? 1.0f : 0.0f;

        float pc = fminf(fmaxf(p, EPSf), 1.0f - EPSf);
        float lg = logf(pc) - log1pf(-pc);
        float u  = noise[b * NL + l];
        float nl = logf(u) - log1pf(-u);
        float x  = lg + nl;                       // TEMP = 1
        float soft = 1.0f / (1.0f + expf(-x));
        float hard = (soft > 0.5f) ? 1.0f : 0.0f; // threshold on pre-mask soft
        hard *= m;
        float bforce = (len < NL && l == len - 1) ? 1.0f : 0.0f;
        if (bforce > hard) hard = bforce;

        float pc2 = fminf(fmaxf(p, 1e-8f), 1.0f - 1e-8f);
        lp   += (hard * logf(pc2) + (1.0f - hard) * log1pf(-pc2)) * m;
        conf += fabsf(p - 0.5f) * m;
        ent  += -(pc2 * logf(pc2) + (1.0f - pc2) * log1pf(-pc2)) * m;

        if (hard > 0.5f) { flags |= (1 << i); cnt++; }
    }

    // ---- exclusive scan of chunk counts (serial; inherently sequential) ----
    sscan[tid] = cnt;
    __syncthreads();
    if (tid == 0) {
        int acc = 0;
        for (int i = 0; i < 256; ++i) { int c = sscan[i]; sscan[i] = acc; acc += c; }
        sscan[256] = acc;
    }
    __syncthreads();
    int base  = sscan[tid];
    int total = sscan[256];

    // ---- record boundary positions ----
    int j = base;
    for (int i = 0; i < TPT; ++i)
        if (flags & (1 << i)) { bpos[b * NL + j] = l0 + i; j++; }

    // ---- shortened attention mask ----
    for (int i = 0; i < TPT; ++i) {
        int l = l0 + i;
        out_sm[b * NL + l] = (l < total) ? 1.0f : 0.0f;
    }

    // ---- reduce stats (wave shfl + 8 partials) ----
    lp = wave_sum32(lp);
    __syncthreads();
    if (lane == 0) spart[wave] = lp;
    __syncthreads();
    if (tid == 0) {
        float s = 0;
#pragma unroll
        for (int w = 0; w < 8; ++w) s += spart[w];
        out_lp[b] = s;
    }
    conf = wave_sum32(conf);
    __syncthreads();
    if (lane == 0) spart[wave] = conf;
    __syncthreads();
    if (tid == 0) {
        float s = 0;
#pragma unroll
        for (int w = 0; w < 8; ++w) s += spart[w];
        out_conf[b] = s / fmaxf((float)len, 1.0f);
    }
    ent = wave_sum32(ent);
    __syncthreads();
    if (lane == 0) spart[wave] = ent;
    __syncthreads();
    if (tid == 0) {
        float s = 0;
#pragma unroll
        for (int w = 0; w < 8; ++w) s += spart[w];
        out_ent[b]   = s;
        counts_ws[b] = total;
        len_ws[b]    = len;
    }
}

// -------------------------------------------------------------------------
// Kernel 3: segment mean pooling. 1 block per (b, s); 128 threads, one
// float4 lane-stream per thread (D/4 = 128 quads).
// -------------------------------------------------------------------------
__global__ __launch_bounds__(128) void k_pool(
    const float* __restrict__ hidden,
    const int* __restrict__ bpos, const int* __restrict__ counts_ws,
    const int* __restrict__ len_ws, float* __restrict__ pooled)
{
    const int b = blockIdx.y;
    const int s = blockIdx.x;
    const int cnt = counts_ws[b];
    const int len = len_ws[b];

    int start, end;
    if (s < cnt) {
        end   = bpos[b * NL + s];
        start = (s == 0) ? 0 : bpos[b * NL + s - 1] + 1;
    } else if (s == cnt) {
        start = (cnt == 0) ? 0 : bpos[b * NL + cnt - 1] + 1;
        end   = len - 1;
    } else {
        start = 1; end = 0;
    }

    v4f* dst4 = (v4f*)(pooled + ((size_t)b * NL + s) * ND);
    const int n = end - start + 1;
    const v4f z = {0.f, 0.f, 0.f, 0.f};
    if (n <= 0) {
        dst4[threadIdx.x] = z;
        return;
    }
    const float inv = 1.0f / ((float)n + 1e-9f);
    const v4f* src4 = (const v4f*)(hidden + ((size_t)b * NL + start) * ND) + threadIdx.x;
    v4f acc = z;
    for (int t = 0; t < n; ++t) acc += src4[(size_t)t * (ND / 4)];
    dst4[threadIdx.x] = acc * inv;
}

// -------------------------------------------------------------------------
// Kernel 4: scalars (loss has weight 0.0, computed faithfully anyway)
// -------------------------------------------------------------------------
__global__ void k_finalize(const int* __restrict__ counts_ws,
                           const int* __restrict__ len_ws,
                           float* __restrict__ out_loss,
                           float* __restrict__ out_nb,
                           float* __restrict__ out_tp)
{
    if (threadIdx.x == 0 && blockIdx.x == 0) {
        float nb = 0.0f, tp = 0.0f;
        for (int b = 0; b < NB; ++b) { nb += (float)counts_ws[b]; tp += (float)len_ws[b]; }
        const float p = 0.2f / (0.2f + 1.0f * 0.8f);   // PRIOR / (PRIOR + sched*(1-PRIOR))
        float n = tp, kk = nb;
        float lpb = lgammaf(n + 1.0f) - lgammaf(kk + 1.0f) - lgammaf(n - kk + 1.0f)
                  + kk * logf(p) + (n - kk) * log1pf(-p);
        *out_loss = (-lpb / n) * 0.0f;                  // BOUNDARY_LOSS_WEIGHT = 0.0
        *out_nb = nb;
        *out_tp = tp;
    }
}

extern "C" void kernel_launch(void* const* d_in, const int* in_sizes, int n_in,
                              void* d_out, int out_size, void* d_ws, size_t ws_size,
                              hipStream_t stream) {
    const float* hidden = (const float*)d_in[0];
    const float* amask  = (const float*)d_in[1];
    const float* noise  = (const float*)d_in[2];
    const float* Wq     = (const float*)d_in[3];
    const float* Wk     = (const float*)d_in[4];

    float* out        = (float*)d_out;
    float* out_pooled = out;                                   // B*L*D
    float* out_loss   = out + (size_t)NB * NL * ND;            // 1
    float* out_nb     = out_loss + 1;                          // 1
    float* out_tp     = out_nb + 1;                            // 1
    float* out_sm     = out_tp + 1;                            // B*L
    float* out_lp     = out_sm + (size_t)NB * NL;              // B
    float* out_conf   = out_lp + NB;                           // B
    float* out_ent    = out_conf + NB;                         // B

    float* ws_cos    = (float*)d_ws;                           // B*L floats
    int*   ws_bpos   = (int*)(ws_cos + (size_t)NB * NL);       // B*L ints
    int*   ws_counts = ws_bpos + (size_t)NB * NL;              // B ints
    int*   ws_len    = ws_counts + NB;                         // B ints

    k_gemm_cos<<<dim3(NL / 16, NB), 256, 0, stream>>>(hidden, Wq, Wk, ws_cos);
    k_boundaries<<<dim3(NB), 256, 0, stream>>>(ws_cos, amask, noise,
                                               out_sm, out_lp, out_conf, out_ent,
                                               ws_bpos, ws_counts, ws_len);
    k_pool<<<dim3(NL, NB), 128, 0, stream>>>(hidden, ws_bpos, ws_counts, ws_len, out_pooled);
    k_finalize<<<1, 1, 0, stream>>>(ws_counts, ws_len, out_loss, out_nb, out_tp);
}
